// ScaleDotAttention_5609227289107
// MI455X (gfx1250) — matrix operand order repaired and checked by reading, not compiled, not run on hardware
//
#include <hip/hip_runtime.h>
#include <hip/hip_bf16.h>
#include <cstdint>

// ---- shapes (fixed by the reference) ----
#define BATCH 16
#define SEQ   2048
#define DMODEL 128
#define QTILE 128     // q rows per block (8 waves x 16)
#define KTILE 64      // k rows per inner tile
#define NKT   (SEQ / KTILE)   // 32
#define LN_EPS 1e-5f

typedef __attribute__((ext_vector_type(16))) __bf16 v16bf;
typedef __attribute__((ext_vector_type(8)))  __bf16 v8bf;
typedef __attribute__((ext_vector_type(8)))  float  v8f;

union V16 { v16bf v; v8bf h[2]; };

// ---- gfx1250 async global->LDS copy (ASYNCcnt-tracked), with safe fallback ----
#if defined(__has_builtin)
#if __has_builtin(__builtin_amdgcn_global_load_async_to_lds_b128)
#define HAVE_ASYNC_LDS 1
#endif
#endif

// GCC-style int4 vector to match the builtin's printed parameter type exactly
typedef int v4i_gcc __attribute__((vector_size(16)));
typedef __attribute__((address_space(1))) v4i_gcc* gptr_v4i;
typedef __attribute__((address_space(3))) v4i_gcc* lptr_v4i;

#define GAS4(p) ((gptr_v4i)(uintptr_t)(p))
#define LAS4(p) ((lptr_v4i)(uint32_t)(uintptr_t)(p))

__device__ __forceinline__ void async_copy16(const float* gsrc, float* ldst) {
#ifdef HAVE_ASYNC_LDS
    __builtin_amdgcn_global_load_async_to_lds_b128(GAS4(gsrc), LAS4(ldst), 0, 0);
#else
    *(float4*)ldst = *(const float4*)gsrc;
#endif
}

__device__ __forceinline__ void async_wait0() {
#ifdef HAVE_ASYNC_LDS
#if __has_builtin(__builtin_amdgcn_s_wait_asynccnt)
    __builtin_amdgcn_s_wait_asynccnt(0);
#else
    asm volatile("s_wait_asynccnt 0" ::: "memory");
#endif
#endif
}

__device__ __forceinline__ v8bf cvt8(const float* p) {
    const float4* p4 = (const float4*)p;
    float4 a = p4[0], b = p4[1];
    v8bf r;
    r[0] = (__bf16)a.x; r[1] = (__bf16)a.y; r[2] = (__bf16)a.z; r[3] = (__bf16)a.w;
    r[4] = (__bf16)b.x; r[5] = (__bf16)b.y; r[6] = (__bf16)b.z; r[7] = (__bf16)b.w;
    return r;
}

__device__ __forceinline__ float rmax16(float x) {
#pragma unroll
    for (int m = 1; m < 16; m <<= 1) x = fmaxf(x, __shfl_xor(x, m, 32));
    return x;
}
__device__ __forceinline__ float rsum16(float x) {
#pragma unroll
    for (int m = 1; m < 16; m <<= 1) x += __shfl_xor(x, m, 32);
    return x;
}

__global__ __launch_bounds__(256, 1)
void flash_attn_ln_kernel(const float* __restrict__ q,
                          const float* __restrict__ k,
                          const float* __restrict__ v,
                          const float* __restrict__ attn_mask,
                          const float* __restrict__ gamma,
                          const float* __restrict__ beta,
                          float* __restrict__ out) {
    // LDS: K [64][128] bf16 | V^T [128][64] bf16 | P stage 8x[16][64] bf16 | mask [128][64] f32
    __shared__ __align__(32) unsigned char lds_raw[81920];
    __bf16* Klds  = (__bf16*)lds_raw;                 // 16 KB
    __bf16* Vtlds = (__bf16*)(lds_raw + 16384);       // 16 KB
    __bf16* Plds  = (__bf16*)(lds_raw + 32768);       // 16 KB
    float*  Mlds  = (float*)(lds_raw + 49152);        // 32 KB

    const int tid  = threadIdx.x;
    const int wave = tid >> 5;
    const int lane = tid & 31;
    const int n    = lane & 15;     // column-in-tile for B/C/D layouts
    const int kh   = lane >> 4;     // K-half / M-half selector

    const int blk = blockIdx.x;
    const int b   = blk >> 4;                 // batch
    const int qt  = blk & 15;                 // q-tile within batch
    const int qwave = qt * QTILE + wave * 16; // this wave's q-row base (within batch)

    // ---- Q fragments (A-layout), resident for the whole kernel ----
    const float* qrow = q + ((size_t)b * SEQ + (qwave + n)) * DMODEL;
    V16 qa[4];
#pragma unroll
    for (int s = 0; s < 4; ++s) {
        const float* qp = qrow + s * 32 + kh * 16;
        qa[s].h[0] = cvt8(qp);
        qa[s].h[1] = cvt8(qp + 8);
    }

    float gscale[8], gbias[8];
#pragma unroll
    for (int dt = 0; dt < 8; ++dt) {
        gscale[dt] = gamma[dt * 16 + n];
        gbias[dt]  = beta[dt * 16 + n];
    }

    // ---- accumulators ----
    v8f O[8];
#pragma unroll
    for (int dt = 0; dt < 8; ++dt) O[dt] = (v8f)0.0f;
    float run_max[8], run_sum[8];
#pragma unroll
    for (int r = 0; r < 8; ++r) { run_max[r] = -__builtin_inff(); run_sum[r] = 0.0f; }

    const int wbase = wave * 1024;  // P-stage base (elements) for this wave

    for (int kb = 0; kb < NKT; ++kb) {
        const int kbase = kb * KTILE;
        __syncthreads();  // previous iteration's LDS readers are done

        // ---- async mask tile copy: [128 q rows][64 k cols] f32, 128B per thread ----
        {
            const float* mbase = attn_mask + (size_t)(qt * QTILE) * SEQ + kbase;
#pragma unroll
            for (int i = 0; i < 8; ++i) {
                int e   = i * 256 + tid;    // 16B chunk index, 0..2047
                int row = e >> 4;           // 16 chunks per 64-col row
                int c4  = (e & 15) * 4;     // float column
                async_copy16(mbase + (size_t)row * SEQ + c4, Mlds + row * 64 + c4);
            }
        }

        // ---- cooperative K/V tile load: f32 -> bf16, V transposed ----
        const float* kptr = k + ((size_t)b * SEQ + kbase) * DMODEL;
        const float* vptr = v + ((size_t)b * SEQ + kbase) * DMODEL;
#pragma unroll
        for (int i = 0; i < 8; ++i) {
            int e   = (i * 256 + tid) * 4;   // 0..8188
            int row = e >> 7;
            int col = e & 127;
            float4 kf = *(const float4*)(kptr + (size_t)row * DMODEL + col);
            __bf16* kd = Klds + row * 128 + col;
            kd[0] = (__bf16)kf.x; kd[1] = (__bf16)kf.y;
            kd[2] = (__bf16)kf.z; kd[3] = (__bf16)kf.w;
            float4 vf = *(const float4*)(vptr + (size_t)row * DMODEL + col);
            Vtlds[(col + 0) * 64 + row] = (__bf16)vf.x;
            Vtlds[(col + 1) * 64 + row] = (__bf16)vf.y;
            Vtlds[(col + 2) * 64 + row] = (__bf16)vf.z;
            Vtlds[(col + 3) * 64 + row] = (__bf16)vf.w;
        }
        async_wait0();
        __syncthreads();

        // prefetch next K/V tile toward L2 while we compute
        if (kb + 1 < NKT) {
            const float* nk = k + ((size_t)b * SEQ + kbase + KTILE) * DMODEL + tid * 32;
            const float* nv = v + ((size_t)b * SEQ + kbase + KTILE) * DMODEL + tid * 32;
            __builtin_prefetch(nk, 0, 1);
            __builtin_prefetch(nv, 0, 1);
        }

        // ---- S = Q * K^T : preload 4 B-fragments per k-step, then 4 WMMA ----
        v8f S[4];
#pragma unroll
        for (int t = 0; t < 4; ++t) S[t] = (v8f)0.0f;
#pragma unroll
        for (int s = 0; s < 4; ++s) {
            v16bf bf[4];
#pragma unroll
            for (int t = 0; t < 4; ++t)
                bf[t] = *(const v16bf*)(Klds + (t * 16 + n) * 128 + s * 32 + kh * 16);
#pragma unroll
            for (int t = 0; t < 4; ++t)
                S[t] = __builtin_amdgcn_wmma_f32_16x16x32_bf16(
                           false, qa[s].v, false, bf[t], (short)0, S[t], false, false);
        }

        // ---- add mask (from LDS), online softmax (row m = 8*kh + r) ----
#pragma unroll
        for (int r = 0; r < 8; ++r) {
            const float* mp = Mlds + (wave * 16 + 8 * kh + r) * 64 + n;
            float mx = -__builtin_inff();
#pragma unroll
            for (int t = 0; t < 4; ++t) {
                S[t][r] += mp[t * 16];
                mx = fmaxf(mx, S[t][r]);
            }
            mx = rmax16(mx);
            float nm   = fmaxf(run_max[r], mx);
            float corr = __expf(run_max[r] - nm);
            run_max[r] = nm;
            float lsum = 0.0f;
#pragma unroll
            for (int t = 0; t < 4; ++t) {
                float p = __expf(S[t][r] - nm);
                S[t][r] = p;
                lsum += p;
            }
            lsum = rsum16(lsum);
            run_sum[r] = run_sum[r] * corr + lsum;
#pragma unroll
            for (int dt = 0; dt < 8; ++dt) O[dt][r] *= corr;
        }

        // ---- stage P (C-layout -> row-major LDS), private per wave ----
#pragma unroll
        for (int r = 0; r < 8; ++r) {
            int row = 8 * kh + r;
#pragma unroll
            for (int t = 0; t < 4; ++t)
                Plds[wbase + row * 64 + t * 16 + n] = (__bf16)S[t][r];
        }

        // ---- O += P * V : preload A-frag + 8 B-fragments per k-step, then 8 WMMA ----
#pragma unroll
        for (int s2 = 0; s2 < 2; ++s2) {
            const __bf16* pb = Plds + wbase + n * 64 + s2 * 32 + kh * 16;
            V16 pa;
            pa.h[0] = *(const v8bf*)pb;
            pa.h[1] = *(const v8bf*)(pb + 8);
            v16bf vf[8];
#pragma unroll
            for (int dt = 0; dt < 8; ++dt)
                vf[dt] = *(const v16bf*)(Vtlds + (dt * 16 + n) * 64 + s2 * 32 + kh * 16);
#pragma unroll
            for (int dt = 0; dt < 8; ++dt)
                O[dt] = __builtin_amdgcn_wmma_f32_16x16x32_bf16(
                            false, pa.v, false, vf[dt], (short)0, O[dt], false, false);
        }
    }

    // ---- finalize: 1/sum, LayerNorm over D=128, gamma/beta, store ----
#pragma unroll
    for (int r = 0; r < 8; ++r) {
        float inv = 1.0f / run_sum[r];
        float sum = 0.0f, sq = 0.0f;
#pragma unroll
        for (int dt = 0; dt < 8; ++dt) {
            float o = O[dt][r] * inv;
            O[dt][r] = o;
            sum += o;
            sq  += o * o;
        }
        sum = rsum16(sum);
        sq  = rsum16(sq);
        float mean = sum * (1.0f / 128.0f);
        float var  = sq * (1.0f / 128.0f) - mean * mean;
        float rstd = rsqrtf(var + LN_EPS);

        int qrow_g = qwave + 8 * kh + r;
        float* orow = out + ((size_t)b * SEQ + qrow_g) * DMODEL;
#pragma unroll
        for (int dt = 0; dt < 8; ++dt) {
            float val = (O[dt][r] - mean) * rstd * gscale[dt] + gbias[dt];
            orow[dt * 16 + n] = val;
        }
    }
}

extern "C" void kernel_launch(void* const* d_in, const int* in_sizes, int n_in,
                              void* d_out, int out_size, void* d_ws, size_t ws_size,
                              hipStream_t stream) {
    const float* q    = (const float*)d_in[0];
    const float* k    = (const float*)d_in[1];
    const float* v    = (const float*)d_in[2];
    const float* mask = (const float*)d_in[3];
    const float* gam  = (const float*)d_in[4];
    const float* bet  = (const float*)d_in[5];
    float* out = (float*)d_out;

    dim3 grid(BATCH * (SEQ / QTILE));  // 256 blocks
    dim3 block(256);                   // 8 wave32
    flash_attn_ln_kernel<<<grid, block, 0, stream>>>(q, k, v, mask, gam, bet, out);
}
